// GAT_Net_Class_48258252538093
// MI455X (gfx1250) — matrix operand order repaired and checked by reading, not compiled, not run on hardware
//
#include <hip/hip_runtime.h>
#include <hip/hip_bf16.h>

#define N_NODES 100000
#define N_EDGES 1600000
#define E_TOT   (N_NODES + N_EDGES)
#define NEG_SLOPE 0.2f

typedef __attribute__((ext_vector_type(16))) __bf16 v16bf;
typedef __attribute__((ext_vector_type(8)))  float  v8f;

// ---------------- CDNA5 async copy (global -> LDS), inline asm ------------------
__device__ __forceinline__ void async_copy_b128(unsigned lds_byte_addr,
                                                const void* gaddr) {
    // VDST = VGPR holding LDS byte address, VADDR = 64-bit global address (GV mode)
    asm volatile("global_load_async_to_lds_b128 %0, %1, off"
                 :: "v"(lds_byte_addr), "v"(gaddr)
                 : "memory");
}
__device__ __forceinline__ void wait_async_zero() {
    asm volatile("s_wait_asynccnt 0x0" ::: "memory");
}

// ---------------- monotonic float<->uint encoding for atomic max ----------------
__device__ __forceinline__ unsigned fenc(float f) {
    unsigned u = __float_as_uint(f);
    return (u & 0x80000000u) ? ~u : (u | 0x80000000u);
}
__device__ __forceinline__ float fdec(unsigned u) {
    return (u & 0x80000000u) ? __uint_as_float(u & 0x7FFFFFFFu)
                             : __uint_as_float(~u);
}

// ---------------- pack W (K x NC, f32 row-major) into bf16 fragments ------------
// Fragment layout: frag f = (kb32 * (NC/16) + ct) * 32 + lane ; 16 contiguous bf16:
//   lane l: col = ct*16 + l%16 ; K = kb32*32 + (l<16?0:16) + j
template<int KDIM, int NC>
__global__ void pack_b_kernel(const float* __restrict__ B, __bf16* __restrict__ P)
{
    constexpr int NFRAG = (KDIM / 32) * (NC / 16) * 32;
    int f = blockIdx.x * blockDim.x + threadIdx.x;
    if (f >= NFRAG) return;
    int lane = f & 31;
    int ct   = (f >> 5) % (NC / 16);
    int kb32 = f / (32 * (NC / 16));
    int col  = ct * 16 + (lane & 15);
    int k0   = kb32 * 32 + ((lane < 16) ? 0 : 16);
    __bf16* dst = P + (size_t)f * 16;
#pragma unroll
    for (int j = 0; j < 16; ++j)
        dst[j] = (__bf16)B[(size_t)(k0 + j) * NC + col];
}

// ---------------- WMMA bf16 GEMM: C[nrows x NC] = A[nrows x K] * B[K x NC] -----
// 8 waves / block, each wave owns a 16-row strip across all NC columns.
// B fragments are async-staged (packed bf16) into LDS, reused by all 8 waves.
template<int KDIM, int NC>
__global__ __launch_bounds__(256)
void gemm_wmma_kernel(const float* __restrict__ A, const __bf16* __restrict__ Bpack,
                      float* __restrict__ C, int nrows)
{
    constexpr int FR_PER_K = (NC / 16) * 32;              // fragments per 32-K step
    constexpr int KSTEPS   = KDIM / 32;
    constexpr int KS_CHUNK = (1024 / FR_PER_K) < KSTEPS ? (1024 / FR_PER_K) : KSTEPS;
    constexpr int FR_CHUNK = KS_CHUNK * FR_PER_K;         // <= 1024 frags = 32 KB

    __shared__ v16bf bsh[FR_CHUNK];

    const int lane = threadIdx.x & 31;
    const int wave = threadIdx.x >> 5;
    const int row0 = (blockIdx.x * 8 + wave) * 16;
    const bool active = row0 < nrows;      // idle waves still hit barriers below

    const int hi  = lane >> 4;             // 0 or 1
    const int l16 = lane & 15;

    v8f acc[NC / 16] = {};

    for (int kc = 0; kc < KSTEPS; kc += KS_CHUNK) {
        // ---- cooperative async stage of packed B fragments into LDS ----
        const int baseFrag = kc * FR_PER_K;
        for (int f = threadIdx.x; f < FR_CHUNK; f += 256) {
            const __bf16* src = Bpack + (size_t)(baseFrag + f) * 16;   // 32 B
            unsigned dst = (unsigned)(size_t)(&bsh[f]);
            async_copy_b128(dst,      src);
            async_copy_b128(dst + 16, src + 8);
        }
        wait_async_zero();
        __syncthreads();

        if (active) {
            for (int ks = 0; ks < KS_CHUNK; ++ks) {
                const int kb = (kc + ks) * 32;
                // A fragment: row = row0+l16; K = kb + hi*16 + {0..15}
                const float4* ap = (const float4*)(A + (size_t)(row0 + l16) * KDIM
                                                     + kb + hi * 16);
                float4 a0 = ap[0], a1 = ap[1], a2 = ap[2], a3 = ap[3];
                float av[16] = { a0.x, a0.y, a0.z, a0.w, a1.x, a1.y, a1.z, a1.w,
                                 a2.x, a2.y, a2.z, a2.w, a3.x, a3.y, a3.z, a3.w };
                v16bf afrag;
#pragma unroll
                for (int i = 0; i < 16; ++i) afrag[i] = (__bf16)av[i];
#pragma unroll
                for (int t = 0; t < NC / 16; ++t) {
                    v16bf bfrag = bsh[ks * FR_PER_K + t * 32 + lane];
                    acc[t] = __builtin_amdgcn_wmma_f32_16x16x32_bf16(
                        false, afrag, false, bfrag, (short)0, acc[t], false, false);
                }
            }
        }
        __syncthreads();
    }

    if (active) {
#pragma unroll
        for (int t = 0; t < NC / 16; ++t) {
            const int col = t * 16 + l16;
#pragma unroll
            for (int v = 0; v < 8; ++v)
                C[(size_t)(row0 + v + hi * 8) * NC + col] = acc[t][v];
        }
    }
}

// ---------------- per-(node,head) attention coefficients -----------------------
template<int H, int CH>
__global__ void alpha_kernel(const float* __restrict__ h,
                             const float* __restrict__ a_src,
                             const float* __restrict__ a_dst,
                             float* __restrict__ as, float* __restrict__ ad)
{
    int idx = blockIdx.x * blockDim.x + threadIdx.x;   // node*H + head
    if (idx >= N_NODES * H) return;
    const int head = idx % H;
    const float* hp = h + (size_t)idx * CH;            // (node*H+head)*CH
    float s = 0.f, d = 0.f;
#pragma unroll
    for (int c = 0; c < CH; ++c) {
        float v = hp[c];
        s += v * a_src[head * CH + c];
        d += v * a_dst[head * CH + c];
    }
    as[idx] = s;
    ad[idx] = d;
}

// ---------------- edge helpers -------------------------------------------------
__device__ __forceinline__ void edge_sd(const int* __restrict__ ei, int e,
                                        int& s, int& d) {
    if (e < N_EDGES) { s = ei[e]; d = ei[N_EDGES + e]; }
    else             { s = d = e - N_EDGES; }          // self loops
}
__device__ __forceinline__ float lrelu(float v) {
    return v > 0.f ? v : NEG_SLOPE * v;
}

template<int H>
__global__ void edge_max_kernel(const int* __restrict__ ei,
                                const float* __restrict__ as,
                                const float* __restrict__ ad,
                                unsigned* __restrict__ m)
{
    int idx = blockIdx.x * blockDim.x + threadIdx.x;
    if (idx >= E_TOT * H) return;
    int e = idx / H, h = idx % H;
    int sN, dN; edge_sd(ei, e, sN, dN);
    float v = lrelu(as[sN * H + h] + ad[dN * H + h]);
    atomicMax(&m[dN * H + h], fenc(v));
}

template<int H>
__global__ void edge_sum_kernel(const int* __restrict__ ei,
                                const float* __restrict__ as,
                                const float* __restrict__ ad,
                                const unsigned* __restrict__ m,
                                float* __restrict__ s)
{
    int idx = blockIdx.x * blockDim.x + threadIdx.x;
    if (idx >= E_TOT * H) return;
    int e = idx / H, h = idx % H;
    int sN, dN; edge_sd(ei, e, sN, dN);
    float v  = lrelu(as[sN * H + h] + ad[dN * H + h]);
    float ex = __expf(v - fdec(m[dN * H + h]));
    atomicAdd(&s[dN * H + h], ex);
}

template<int H, int CH>
__global__ void edge_msg_kernel(const int* __restrict__ ei,
                                const float* __restrict__ as,
                                const float* __restrict__ ad,
                                const unsigned* __restrict__ m,
                                const float* __restrict__ s,
                                const float* __restrict__ hfeat,
                                float* __restrict__ out)
{
    int idx = blockIdx.x * blockDim.x + threadIdx.x;
    if (idx >= E_TOT * H) return;
    int e = idx / H, h = idx % H;
    int sN, dN; edge_sd(ei, e, sN, dN);
    float v = lrelu(as[sN * H + h] + ad[dN * H + h]);
    float w = __expf(v - fdec(m[dN * H + h])) / (s[dN * H + h] + 1e-16f);
    const float4* hs = (const float4*)(hfeat + ((size_t)sN * H + h) * CH);
    float*        op = out + ((size_t)dN * H + h) * CH;
#pragma unroll
    for (int c4 = 0; c4 < CH / 4; ++c4) {
        float4 hv = hs[c4];
        atomicAdd(&op[c4 * 4 + 0], hv.x * w);
        atomicAdd(&op[c4 * 4 + 1], hv.y * w);
        atomicAdd(&op[c4 * 4 + 2], hv.z * w);
        atomicAdd(&op[c4 * 4 + 3], hv.w * w);
    }
}

// ---------------- misc elementwise kernels -------------------------------------
__global__ void zero_kernel(unsigned* __restrict__ p, int n) {
    int i = blockIdx.x * blockDim.x + threadIdx.x;
    if (i < n) p[i] = 0u;
}

__global__ void bias_elu_kernel(float* __restrict__ h, const float* __restrict__ b,
                                int total, int F)
{
    int i = blockIdx.x * blockDim.x + threadIdx.x;
    if (i >= total) return;
    float v = h[i] + b[i % F];
    h[i] = v > 0.f ? v : (__expf(v) - 1.f);
}

__global__ void logsoftmax_kernel(const float* __restrict__ in,
                                  const float* __restrict__ b,
                                  float* __restrict__ out)
{
    int node = blockIdx.x * blockDim.x + threadIdx.x;
    if (node >= N_NODES) return;
    float v[32];
    float mx = -3.4e38f;
#pragma unroll
    for (int j = 0; j < 32; ++j) {
        v[j] = in[(size_t)node * 32 + j] + b[j];
        mx = fmaxf(mx, v[j]);
    }
    float sum = 0.f;
#pragma unroll
    for (int j = 0; j < 32; ++j) sum += __expf(v[j] - mx);
    float ls = __logf(sum);
#pragma unroll
    for (int j = 0; j < 32; ++j)
        out[(size_t)node * 32 + j] = v[j] - mx - ls;
}

// ---------------- host orchestration -------------------------------------------
extern "C" void kernel_launch(void* const* d_in, const int* in_sizes, int n_in,
                              void* d_out, int out_size, void* d_ws, size_t ws_size,
                              hipStream_t stream) {
    (void)in_sizes; (void)n_in; (void)out_size; (void)ws_size;

    const float* x   = (const float*)d_in[0];
    const int*   ei  = (const int*)  d_in[1];
    const float* W1  = (const float*)d_in[2];
    const float* aS1 = (const float*)d_in[3];
    const float* aD1 = (const float*)d_in[4];
    const float* b1  = (const float*)d_in[5];
    const float* W2  = (const float*)d_in[6];
    const float* aS2 = (const float*)d_in[7];
    const float* aD2 = (const float*)d_in[8];
    const float* b2  = (const float*)d_in[9];

    const int N = N_NODES;
    float* ws = (float*)d_ws;
    float*    h1   = ws;                       // N*64
    float*    as1  = h1  + (size_t)N * 64;     // N*8
    float*    ad1  = as1 + (size_t)N * 8;      // N*8
    unsigned* m1   = (unsigned*)(ad1 + (size_t)N * 8);   // N*8
    float*    s1   = (float*)m1 + (size_t)N * 8;         // N*8
    float*    out1 = s1  + (size_t)N * 8;      // N*64 (becomes layer-2 input)
    float*    h2   = out1 + (size_t)N * 64;    // N*32
    float*    as2  = h2  + (size_t)N * 32;     // N
    float*    ad2  = as2 + (size_t)N;          // N
    unsigned* m2   = (unsigned*)(ad2 + (size_t)N);       // N
    float*    s2   = (float*)m2 + (size_t)N;             // N
    float*    out2 = s2  + (size_t)N;          // N*32
    __bf16*   bp1  = (__bf16*)(out2 + (size_t)N * 32);   // 512*64 packed bf16
    __bf16*   bp2  = bp1 + (size_t)512 * 64;             // 64*32 packed bf16

    auto cdiv = [](long long a, long long b) { return (int)((a + b - 1) / b); };
    const int T = 256;
    const int gemmBlocks = cdiv(N, 16 * 8);    // 8 waves/block, 16 rows/wave
    const int EH8 = E_TOT * 8;

    // ---------------- weight packing (f32 -> bf16 WMMA fragments) -------------
    pack_b_kernel<512, 64><<<cdiv(16 * 4 * 32, T), T, 0, stream>>>(W1, bp1);
    pack_b_kernel<64, 32><<<cdiv(2 * 2 * 32, T), T, 0, stream>>>(W2, bp2);

    // ---------------- layer 1 ----------------
    gemm_wmma_kernel<512, 64><<<gemmBlocks, T, 0, stream>>>(x, bp1, h1, N);
    alpha_kernel<8, 8><<<cdiv((long long)N * 8, T), T, 0, stream>>>(h1, aS1, aD1, as1, ad1);
    // zero m1|s1|out1 (contiguous: N*8 + N*8 + N*64 words)
    zero_kernel<<<cdiv((long long)N * 80, T), T, 0, stream>>>(m1, N * 80);
    edge_max_kernel<8><<<cdiv(EH8, T), T, 0, stream>>>(ei, as1, ad1, m1);
    edge_sum_kernel<8><<<cdiv(EH8, T), T, 0, stream>>>(ei, as1, ad1, m1, s1);
    edge_msg_kernel<8, 8><<<cdiv(EH8, T), T, 0, stream>>>(ei, as1, ad1, m1, s1, h1, out1);
    bias_elu_kernel<<<cdiv((long long)N * 64, T), T, 0, stream>>>(out1, b1, N * 64, 64);

    // ---------------- layer 2 ----------------
    gemm_wmma_kernel<64, 32><<<gemmBlocks, T, 0, stream>>>(out1, bp2, h2, N);
    alpha_kernel<1, 32><<<cdiv(N, T), T, 0, stream>>>(h2, aS2, aD2, as2, ad2);
    // zero m2|s2|out2 (contiguous: N + N + N*32 words)
    zero_kernel<<<cdiv((long long)N * 34, T), T, 0, stream>>>(m2, N * 34);
    edge_max_kernel<1><<<cdiv(E_TOT, T), T, 0, stream>>>(ei, as2, ad2, m2);
    edge_sum_kernel<1><<<cdiv(E_TOT, T), T, 0, stream>>>(ei, as2, ad2, m2, s2);
    edge_msg_kernel<1, 32><<<cdiv(E_TOT, T), T, 0, stream>>>(ei, as2, ad2, m2, s2, h2, out2);

    logsoftmax_kernel<<<cdiv(N, T), T, 0, stream>>>(out2, b2, (float*)d_out);
}
